// RelationalMemoryCell_47442208751989
// MI455X (gfx1250) — matrix operand order repaired and checked by reading, not compiled, not run on hardware
//
#include <hip/hip_runtime.h>

// ---------------- types for CDNA5 WMMA ----------------
typedef __attribute__((ext_vector_type(16))) __bf16        v16bf;
typedef __attribute__((ext_vector_type(8)))  float         v8f;
typedef __attribute__((ext_vector_type(4)))  unsigned int  v4u;

// ---------------- problem constants ----------------
#define B_SZ     2048
#define S_SZ     8
#define H_SZ     8
#define HS_SZ    128
#define M_SZ     1024     // H*HS
#define QKV_SZ   384      // 2*KS + VS
#define TOTAL_SZ 3072     // QKV*H
#define IN_DIM   1024
#define SP1      9        // S+1
#define ROWS     (B_SZ * SP1)    // 18432
#define GROWS    (B_SZ * S_SZ)   // 16384

// ---------------- helpers ----------------
__device__ __forceinline__ unsigned short f2bf(float f) {
  unsigned int u = __float_as_uint(f);
  u += 0x7fffu + ((u >> 16) & 1u);      // round to nearest even
  return (unsigned short)(u >> 16);
}
__device__ __forceinline__ float bf2f(unsigned short s) {
  return __uint_as_float(((unsigned int)s) << 16);
}
__device__ __forceinline__ unsigned int pack2(float a, float b) {
  return (unsigned int)f2bf(a) | ((unsigned int)f2bf(b) << 16);
}

// CDNA5 async global->LDS copy (16B per lane), tracked by ASYNCcnt.
__device__ __forceinline__ void async_copy_b128(unsigned lds_addr, const void* gptr) {
  asm volatile("global_load_async_to_lds_b128 %0, %1, off"
               :: "v"(lds_addr), "v"((unsigned long long)gptr)
               : "memory");
}
__device__ __forceinline__ void wait_async0() {
  asm volatile("s_wait_asynccnt 0x0" ::: "memory");
}

// ---------------- WMMA GEMM ----------------
// D = A(bf16)[M,K] @ Bt(bf16)[N,K]^T + bias, templated epilogue.
// 128x128 macro-tile, 8 waves, wave tile 64x32 (4x2 frags), BK=64 (2 sub-steps).
// Double-buffered LDS filled by GLOBAL_LOAD_ASYNC_TO_LDS_B128, overlapped with WMMA.
#define BM   128
#define BN   128
#define BKK  64
#define LDSS 72   // padded row stride in bf16 (144B = 36 dwords -> conflict-free frags)

union FragU { v4u u[2]; v16bf f; };

template <bool HASADD, bool RELU, bool OUTF32, bool OUTBF16>
__global__ __launch_bounds__(256)
void gemm_bf16(const unsigned short* __restrict__ A,    // [M,K] bf16 row-major
               const unsigned short* __restrict__ Bt,   // [N,K] bf16 (pre-transposed weights)
               const float* __restrict__ bias,
               const float* __restrict__ Cadd,
               float* __restrict__ Df32,
               unsigned short* __restrict__ Dbf16,
               int M, int N, int K)
{
  __shared__ unsigned short As[2][BM * LDSS];
  __shared__ unsigned short Bs[2][BN * LDSS];

  const int tid  = threadIdx.x;
  const int lane = tid & 31;
  const int wave = tid >> 5;
  const int wm   = (wave >> 2) * 64;
  const int wn   = (wave & 3) * 32;
  const int m0   = blockIdx.y * BM;
  const int n0   = blockIdx.x * BN;
  const int fr   = lane & 15;      // fragment row/col within 16
  const int kh   = lane >> 4;      // K-half select per ISA layout

  // staging map: 1024 slots of 8 bf16; thread owns rows (tid>>3)+32*it, seg (tid&7)*8
  const int srow = tid >> 3;
  const int sq   = (tid & 7) * 8;

  // LDS byte addresses for async DMA (flat-pointer low 32 bits = LDS address, ISA 10.2)
  const unsigned asBase = (unsigned)(unsigned long long)&As[0][0] + (unsigned)(srow * LDSS + sq) * 2u;
  const unsigned bsBase = (unsigned)(unsigned long long)&Bs[0][0] + (unsigned)(srow * LDSS + sq) * 2u;

  v8f acc[4][2];
  #pragma unroll
  for (int i = 0; i < 4; ++i)
    #pragma unroll
    for (int j = 0; j < 2; ++j)
      #pragma unroll
      for (int e = 0; e < 8; ++e) acc[i][j][e] = 0.f;

  auto issue_tile = [&](int kt, int buf) {
    const unsigned short* pA = A  + (size_t)(m0 + srow) * K + kt + sq;
    const unsigned short* pB = Bt + (size_t)(n0 + srow) * K + kt + sq;
    const unsigned la = asBase + (unsigned)buf * (BM * LDSS * 2u);
    const unsigned lb = bsBase + (unsigned)buf * (BN * LDSS * 2u);
    #pragma unroll
    for (int it = 0; it < 4; ++it) {
      async_copy_b128(la + (unsigned)it * (32u * LDSS * 2u), pA + (size_t)(32 * it) * K);
      async_copy_b128(lb + (unsigned)it * (32u * LDSS * 2u), pB + (size_t)(32 * it) * K);
    }
  };

  issue_tile(0, 0);
  int cur = 0;
  for (int kt = 0; kt < K; kt += BKK) {
    wait_async0();          // this wave's tile-cur DMA complete
    __syncthreads();        // all waves' DMA complete -> buf[cur] readable
    if (kt + BKK < K) issue_tile(kt + BKK, cur ^ 1);   // fill other buffer during compute

    #pragma unroll
    for (int ks = 0; ks < 2; ++ks) {
      const int kso = ks * 32;
      FragU a[4], b[2];
      #pragma unroll
      for (int mi = 0; mi < 4; ++mi) {
        const unsigned short* p = &As[cur][(wm + mi * 16 + fr) * LDSS + kso + kh * 8];
        a[mi].u[0] = *(const v4u*)p;          // K = kso + kh*8 .. +7
        a[mi].u[1] = *(const v4u*)(p + 16);   // K = kso + 16 + kh*8 .. +7
      }
      #pragma unroll
      for (int ni = 0; ni < 2; ++ni) {
        const unsigned short* p = &Bs[cur][(wn + ni * 16 + fr) * LDSS + kso + kh * 8];
        b[ni].u[0] = *(const v4u*)p;          // K = kso + kh*8 .. +7
        b[ni].u[1] = *(const v4u*)(p + 16);   // K = kso + 16 + kh*8 .. +7
      }
      #pragma unroll
      for (int mi = 0; mi < 4; ++mi)
        #pragma unroll
        for (int ni = 0; ni < 2; ++ni)
          acc[mi][ni] = __builtin_amdgcn_wmma_f32_16x16x32_bf16(
              false, a[mi].f, false, b[ni].f, (short)0, acc[mi][ni], false, false);
    }
    __syncthreads();        // reads of buf[cur] done before it is refilled next round
    cur ^= 1;
  }

  // ---- straight-line templated epilogue ----
  #pragma unroll
  for (int mi = 0; mi < 4; ++mi) {
    #pragma unroll
    for (int ni = 0; ni < 2; ++ni) {
      int col  = n0 + wn + ni * 16 + fr;
      int rowb = m0 + wm + mi * 16 + kh * 8;
      float bv = bias[col];
      #pragma unroll
      for (int i = 0; i < 8; ++i) {
        float vv = acc[mi][ni][i] + bv;
        size_t off = (size_t)(rowb + i) * N + col;
        if (HASADD)  vv += Cadd[off];
        if (RELU)    vv = fmaxf(vv, 0.f);
        if (OUTF32)  Df32[off] = vv;
        if (OUTBF16) Dbf16[off] = f2bf(vv);
      }
    }
  }
}

// ---------------- attention: per (b,h), 9x9 softmax over 128-d heads ----------------
__global__ __launch_bounds__(128)
void attn_kernel(const unsigned short* __restrict__ qkv, float* __restrict__ att)
{
  __shared__ float q[SP1][HS_SZ], k[SP1][HS_SZ], v[SP1][HS_SZ];
  __shared__ float w[SP1][SP1];
  const int t  = threadIdx.x;           // 0..127 (d index)
  const int bh = blockIdx.x;
  const int b  = bh >> 3, h = bh & 7;
  const float scale = 0.05103103630798288f;   // 384^-0.5 (QKV^-0.5 per reference)

  for (int s = 0; s < SP1; ++s) {
    size_t base = (size_t)(b * SP1 + s) * TOTAL_SZ + h * QKV_SZ;
    q[s][t] = bf2f(qkv[base + t]) * scale;
    k[s][t] = bf2f(qkv[base + 128 + t]);
    v[s][t] = bf2f(qkv[base + 256 + t]);
  }
  __syncthreads();
  if (t < 81) {
    int i = t / 9, j = t % 9;
    float s = 0.f;
    for (int d = 0; d < HS_SZ; ++d) s += q[i][d] * k[j][d];
    w[i][j] = s;
  }
  __syncthreads();
  if (t < 9) {
    float mx = w[t][0];
    for (int j = 1; j < 9; ++j) mx = fmaxf(mx, w[t][j]);
    float e[9], sum = 0.f;
    for (int j = 0; j < 9; ++j) { e[j] = __expf(w[t][j] - mx); sum += e[j]; }
    float inv = 1.f / sum;
    for (int j = 0; j < 9; ++j) w[t][j] = e[j] * inv;
  }
  __syncthreads();
  for (int i = 0; i < SP1; ++i) {
    float s = 0.f;
    for (int j = 0; j < SP1; ++j) s += w[i][j] * v[j][t];
    att[(size_t)(b * SP1 + i) * M_SZ + h * HS_SZ + t] = s;
  }
}

// ---------------- conversion kernels ----------------
__global__ void cvt_bf16_v4(const float* __restrict__ src, unsigned short* __restrict__ dst, int nq) {
  int i = blockIdx.x * blockDim.x + threadIdx.x;
  if (i >= nq) return;
  float4 v = *(const float4*)(src + (size_t)i * 4);
  uint2 o; o.x = pack2(v.x, v.y); o.y = pack2(v.z, v.w);
  *(uint2*)(dst + (size_t)i * 4) = o;
}

// f32 [K,N] -> bf16 [N,K] tiled transpose (32x32 tiles via LDS)
__global__ __launch_bounds__(256)
void cvt_transpose_bf16(const float* __restrict__ src, unsigned short* __restrict__ dst,
                        int K, int N) {
  __shared__ float tile[32][33];
  const int k0 = blockIdx.y * 32, n0 = blockIdx.x * 32;
  const int tx = threadIdx.x, ty = threadIdx.y;   // 32 x 8
  #pragma unroll
  for (int i = 0; i < 32; i += 8)
    tile[ty + i][tx] = src[(size_t)(k0 + ty + i) * N + n0 + tx];
  __syncthreads();
  #pragma unroll
  for (int i = 0; i < 32; i += 8)
    dst[(size_t)(n0 + ty + i) * K + k0 + tx] = f2bf(tile[tx][ty + i]);
}

// ---------------- elementwise kernels (x4 vectorized) ----------------
__global__ void build_mpi(const float* __restrict__ memory, const float* __restrict__ x,
                          unsigned short* __restrict__ mpi, int nq) {
  int i = blockIdx.x * blockDim.x + threadIdx.x;  // quads
  if (i >= nq) return;
  int row = i >> 8;                 // 256 quads per 1024-row
  int mq  = (i & 255) << 2;
  int b = row / 9, s = row - b * 9;
  const float* src = (s < 8) ? (memory + (size_t)(b * 8 + s) * 1024 + mq)
                             : (x + (size_t)b * 1024 + mq);
  float4 v = *(const float4*)src;
  uint2 o; o.x = pack2(v.x, v.y); o.y = pack2(v.z, v.w);
  *(uint2*)(mpi + (size_t)i * 4) = o;
}

__global__ void m1_kernel(const float* __restrict__ memory, const float* __restrict__ x,
                          float* __restrict__ att_m1, unsigned short* __restrict__ m1bf, int nq) {
  int i = blockIdx.x * blockDim.x + threadIdx.x;  // quads
  if (i >= nq) return;
  int row = i >> 8;
  int mq  = (i & 255) << 2;
  int b = row / 9, s = row - b * 9;
  const float* src = (s < 8) ? (memory + (size_t)(b * 8 + s) * 1024 + mq)
                             : (x + (size_t)b * 1024 + mq);
  float4 base = *(const float4*)src;
  float4 av   = *(const float4*)(att_m1 + (size_t)i * 4);
  float4 v;
  v.x = base.x + av.x; v.y = base.y + av.y; v.z = base.z + av.z; v.w = base.w + av.w;
  *(float4*)(att_m1 + (size_t)i * 4) = v;        // in place: att -> m1 (f32)
  uint2 o; o.x = pack2(v.x, v.y); o.y = pack2(v.z, v.w);
  *(uint2*)(m1bf + (size_t)i * 4) = o;
}

// ---------------- gate GEMVs: one wave per row ----------------
__global__ __launch_bounds__(256)
void gi_kernel(const float* __restrict__ x, const float* __restrict__ kgi,
               const float* __restrict__ bgi, float* __restrict__ gi) {
  int wid  = (blockIdx.x * blockDim.x + threadIdx.x) >> 5;
  int lane = threadIdx.x & 31;
  if (wid >= B_SZ) return;
  const float* rowp = x + (size_t)wid * M_SZ;
  float a0 = 0.f, a1 = 0.f;
  for (int kk = lane; kk < M_SZ; kk += 32) {
    float t = rowp[kk];
    a0 += t * kgi[kk * 2];
    a1 += t * kgi[kk * 2 + 1];
  }
  #pragma unroll
  for (int o = 16; o > 0; o >>= 1) { a0 += __shfl_down(a0, o); a1 += __shfl_down(a1, o); }
  if (lane == 0) {
    gi[wid * 2 + 0] = a0 + bgi[0];
    gi[wid * 2 + 1] = a1 + bgi[1];
  }
}

__global__ __launch_bounds__(256)
void gates_kernel(const float* __restrict__ memory, const float* __restrict__ kgm,
                  const float* __restrict__ bgm, const float* __restrict__ gi,
                  float* __restrict__ gates) {
  int wid  = (blockIdx.x * blockDim.x + threadIdx.x) >> 5;
  int lane = threadIdx.x & 31;
  if (wid >= GROWS) return;
  const float* rowp = memory + (size_t)wid * M_SZ;
  float a0 = 0.f, a1 = 0.f;
  for (int kk = lane; kk < M_SZ; kk += 32) {
    float t = tanhf(rowp[kk]);
    a0 += t * kgm[kk * 2];
    a1 += t * kgm[kk * 2 + 1];
  }
  #pragma unroll
  for (int o = 16; o > 0; o >>= 1) { a0 += __shfl_down(a0, o); a1 += __shfl_down(a1, o); }
  if (lane == 0) {
    int b = wid >> 3;
    gates[wid * 2 + 0] = a0 + bgm[0] + gi[b * 2 + 0];
    gates[wid * 2 + 1] = a1 + bgm[1] + gi[b * 2 + 1];
  }
}

// ---------------- final combine (x4 vectorized) ----------------
__global__ void final_kernel(const float* __restrict__ m2, const float* __restrict__ gates,
                             const float* __restrict__ memory, float* __restrict__ out, int nq) {
  int i = blockIdx.x * blockDim.x + threadIdx.x;  // quads
  if (i >= nq) return;
  int b   = i >> 11;              // 2048 quads per batch (8192 elems)
  int rem = i & 2047;
  int s   = rem >> 8;
  int mq  = (rem & 255) << 2;
  float g0 = gates[(b * 8 + s) * 2 + 0];
  float g1 = gates[(b * 8 + s) * 2 + 1];
  float ig = 1.f / (1.f + __expf(-g0));            // INPUT_BIAS = 0
  float fg = 1.f / (1.f + __expf(-(g1 + 1.0f)));   // FORGET_BIAS = 1
  float4 nx = *(const float4*)(m2 + (size_t)(b * 9 + s) * 1024 + mq);
  float4 mm = *(const float4*)(memory + (size_t)i * 4);
  float4 o;
  o.x = ig * tanhf(nx.x) + fg * mm.x;
  o.y = ig * tanhf(nx.y) + fg * mm.y;
  o.z = ig * tanhf(nx.z) + fg * mm.z;
  o.w = ig * tanhf(nx.w) + fg * mm.w;
  *(float4*)(out + (size_t)i * 4) = o;
}

// ---------------- launcher ----------------
extern "C" void kernel_launch(void* const* d_in, const int* in_sizes, int n_in,
                              void* d_out, int out_size, void* d_ws, size_t ws_size,
                              hipStream_t stream) {
  (void)in_sizes; (void)n_in; (void)out_size; (void)ws_size;
  const float* inputs = (const float*)d_in[0];
  const float* memory = (const float*)d_in[1];
  const float* k_qkv  = (const float*)d_in[2];
  const float* b_qkv  = (const float*)d_in[3];
  const float* k_gi   = (const float*)d_in[4];
  const float* b_gi   = (const float*)d_in[5];
  const float* k_gm   = (const float*)d_in[6];
  const float* b_gm   = (const float*)d_in[7];
  const float* k_in   = (const float*)d_in[8];
  const float* b_in   = (const float*)d_in[9];
  const float* mlp_k0 = (const float*)d_in[10];
  const float* mlp_b0 = (const float*)d_in[11];
  const float* mlp_k1 = (const float*)d_in[12];
  const float* mlp_b1 = (const float*)d_in[13];

  char* ws = (char*)d_ws;
  size_t off = 0;
  auto alloc = [&](size_t bytes) -> void* {
    void* p = ws + off;
    off += (bytes + 255) & ~(size_t)255;
    return p;
  };
  unsigned short* in_bf   = (unsigned short*)alloc((size_t)B_SZ * IN_DIM * 2);
  unsigned short* win_bf  = (unsigned short*)alloc((size_t)IN_DIM * M_SZ * 2);   // [N,K]
  unsigned short* wqkv_bf = (unsigned short*)alloc((size_t)M_SZ * TOTAL_SZ * 2); // [N,K]
  unsigned short* wk0_bf  = (unsigned short*)alloc((size_t)M_SZ * M_SZ * 2);     // [N,K]
  unsigned short* wk1_bf  = (unsigned short*)alloc((size_t)M_SZ * M_SZ * 2);     // [N,K]
  float*          x_f32   = (float*)alloc((size_t)B_SZ * M_SZ * 4);
  unsigned short* mpi_bf  = (unsigned short*)alloc((size_t)ROWS * M_SZ * 2);
  unsigned short* qkv_bf  = (unsigned short*)alloc((size_t)ROWS * TOTAL_SZ * 2);
  float*          att_m1  = (float*)alloc((size_t)ROWS * M_SZ * 4);   // att, then m1 (f32)
  unsigned short* m1_bf   = (unsigned short*)alloc((size_t)ROWS * M_SZ * 2);
  unsigned short* h_bf    = (unsigned short*)alloc((size_t)ROWS * M_SZ * 2);
  float*          m2_f32  = (float*)alloc((size_t)ROWS * M_SZ * 4);
  float*          gi_buf  = (float*)alloc((size_t)B_SZ * 2 * 4);
  float*          gates_b = (float*)alloc((size_t)GROWS * 2 * 4);

  {
    int nq = B_SZ * IN_DIM / 4;
    cvt_bf16_v4<<<(nq + 255) / 256, 256, 0, stream>>>(inputs, in_bf, nq);
  }
  // weights: convert + transpose to [N,K] so GEMM B-staging is contiguous b128 DMAs
  cvt_transpose_bf16<<<dim3(M_SZ / 32, IN_DIM / 32), dim3(32, 8), 0, stream>>>(k_in, win_bf, IN_DIM, M_SZ);
  cvt_transpose_bf16<<<dim3(TOTAL_SZ / 32, M_SZ / 32), dim3(32, 8), 0, stream>>>(k_qkv, wqkv_bf, M_SZ, TOTAL_SZ);
  cvt_transpose_bf16<<<dim3(M_SZ / 32, M_SZ / 32), dim3(32, 8), 0, stream>>>(mlp_k0, wk0_bf, M_SZ, M_SZ);
  cvt_transpose_bf16<<<dim3(M_SZ / 32, M_SZ / 32), dim3(32, 8), 0, stream>>>(mlp_k1, wk1_bf, M_SZ, M_SZ);

  // x = inputs @ kernel_in + bias_in  (f32 out)
  gemm_bf16<false, false, true, false><<<dim3(M_SZ / BN, B_SZ / BM), 256, 0, stream>>>(
      in_bf, win_bf, b_in, nullptr, x_f32, nullptr, B_SZ, M_SZ, IN_DIM);

  build_mpi<<<(ROWS * M_SZ / 4 + 255) / 256, 256, 0, stream>>>(memory, x_f32, mpi_bf, ROWS * M_SZ / 4);

  // qkv = mpi @ kernel_qkv + bias_qkv  (bf16 out for attention)
  gemm_bf16<false, false, false, true><<<dim3(TOTAL_SZ / BN, ROWS / BM), 256, 0, stream>>>(
      mpi_bf, wqkv_bf, b_qkv, nullptr, nullptr, qkv_bf, ROWS, TOTAL_SZ, M_SZ);

  attn_kernel<<<B_SZ * H_SZ, 128, 0, stream>>>(qkv_bf, att_m1);

  m1_kernel<<<(ROWS * M_SZ / 4 + 255) / 256, 256, 0, stream>>>(memory, x_f32, att_m1, m1_bf, ROWS * M_SZ / 4);

  // h = relu(m1 @ mlp_k0 + b0)  (bf16 out)
  gemm_bf16<false, true, false, true><<<dim3(M_SZ / BN, ROWS / BM), 256, 0, stream>>>(
      m1_bf, wk0_bf, mlp_b0, nullptr, nullptr, h_bf, ROWS, M_SZ, M_SZ);

  // m2 = m1 + h @ mlp_k1 + b1  (f32 out, fused residual)
  gemm_bf16<true, false, true, false><<<dim3(M_SZ / BN, ROWS / BM), 256, 0, stream>>>(
      h_bf, wk1_bf, mlp_b1, att_m1, m2_f32, nullptr, ROWS, M_SZ, M_SZ);

  gi_kernel<<<(B_SZ * 32 + 255) / 256, 256, 0, stream>>>(x_f32, k_gi, b_gi, gi_buf);
  gates_kernel<<<(GROWS * 32 + 255) / 256, 256, 0, stream>>>(memory, k_gm, b_gm, gi_buf, gates_b);
  final_kernel<<<(B_SZ * S_SZ * M_SZ / 4 + 255) / 256, 256, 0, stream>>>(
      m2_f32, gates_b, memory, (float*)d_out, B_SZ * S_SZ * M_SZ / 4);
}
